// Generator_50964081935512
// MI455X (gfx1250) — matrix operand order repaired and checked by reading, not compiled, not run on hardware
//
#include <hip/hip_runtime.h>
#include <hip/hip_bf16.h>
#include <cmath>

typedef __attribute__((ext_vector_type(16))) _Float16 v16h;
typedef __attribute__((ext_vector_type(8)))  _Float16 v8h;
typedef __attribute__((ext_vector_type(8)))  float    v8f;

#define SQRT2F 1.41421356237309515f

// ---------------------------------------------------------------------------
// Mapping network: PixelNorm + 8 x EqualLinear(lr_mul=0.01) + fused lrelu.
// One block of 512 threads; x[4][512] lives in LDS. Tiny (17 MFLOP) -> scalar.
// ---------------------------------------------------------------------------
__global__ void __launch_bounds__(512) k_mapping(const float* __restrict__ z,
                                                 const float* __restrict__ mw,
                                                 const float* __restrict__ mb,
                                                 float* __restrict__ wlat) {
  __shared__ float xs[4][512];
  __shared__ float red[512];
  const int t = threadIdx.x;
  for (int b = 0; b < 4; ++b) {
    float v = z[b * 512 + t];
    red[t] = v * v;
    __syncthreads();
    for (int off = 256; off > 0; off >>= 1) {
      if (t < off) red[t] += red[t + off];
      __syncthreads();
    }
    float inv = rsqrtf(red[0] * (1.0f / 512.0f) + 1e-8f);
    __syncthreads();
    xs[b][t] = v * inv;
  }
  __syncthreads();
  const float scale = 0.01f / 22.6274169979695223f;  // lr_mul / sqrt(512)
  for (int l = 0; l < 8; ++l) {
    float acc0 = 0.f, acc1 = 0.f, acc2 = 0.f, acc3 = 0.f;
    const float* Wl = mw + (l * 512 + t) * 512;
    for (int i = 0; i < 512; ++i) {
      float w = Wl[i] * scale;
      acc0 += xs[0][i] * w; acc1 += xs[1][i] * w;
      acc2 += xs[2][i] * w; acc3 += xs[3][i] * w;
    }
    float bias = mb[l * 512 + t] * 0.01f;
    __syncthreads();
    float a[4] = {acc0, acc1, acc2, acc3};
    for (int b = 0; b < 4; ++b) {
      float v = a[b] + bias;
      xs[b][t] = (v > 0.f ? v : 0.2f * v) * SQRT2F;
    }
    __syncthreads();
  }
  for (int b = 0; b < 4; ++b) wlat[b * 512 + t] = xs[b][t];
}

// ---------------------------------------------------------------------------
// style[b,i] = wlat[b,:] . mw[i,:] / sqrt(512) + mb[i]
// ---------------------------------------------------------------------------
__global__ void k_style(const float* __restrict__ wlat, const float* __restrict__ mw,
                        const float* __restrict__ mb, float* __restrict__ style, int I) {
  int idx = blockIdx.x * 256 + threadIdx.x;
  if (idx >= 4 * I) return;
  int b = idx / I, i = idx - b * I;
  float acc = 0.f;
  const float* wr = mw + i * 512;
  const float* wl = wlat + b * 512;
  for (int j = 0; j < 512; ++j) acc += wl[j] * wr[j];
  style[idx] = acc * 0.04419417382415922f + mb[i];  // 1/sqrt(512)
}

// ---------------------------------------------------------------------------
// demod[b,o] = rsqrt( sum_i style[b,i]^2 * sum_k (W[o,i,k]*scale)^2 + 1e-8 )
// scale^2 = 1/(I*9)
// ---------------------------------------------------------------------------
__global__ void k_demod(const float* __restrict__ style, const float* __restrict__ W,
                        float* __restrict__ demod, int I, int O) {
  int idx = blockIdx.x * 256 + threadIdx.x;
  if (idx >= 4 * O) return;
  int b = idx / O, o = idx - b * O;
  const float sc2 = 1.0f / (float)(I * 9);
  const float* Wo = W + o * I * 9;
  const float* st = style + b * I;
  float acc = 0.f;
  for (int i = 0; i < I; ++i) {
    float w2 = 0.f;
    for (int k = 0; k < 9; ++k) { float w = Wo[i * 9 + k]; w2 += w * w; }
    float s = st[i];
    acc += s * s * w2;
  }
  demod[idx] = rsqrtf(acc * sc2 + 1e-8f);
}

// ---------------------------------------------------------------------------
// Convert conv weight to f16, laid out [tap][I][O] (tap-flipped for transposed
// conv). Wh[t,i,o] = W[o,i, flip? 8-t : t] * scale.
// ---------------------------------------------------------------------------
__global__ void k_wconv(const float* __restrict__ W, _Float16* __restrict__ Wh,
                        int O, int I, int flip, float scale) {
  int idx = blockIdx.x * 256 + threadIdx.x;
  if (idx >= O * I * 9) return;
  int t = idx % 9; int tmp = idx / 9;
  int i = tmp % I; int o = tmp / I;
  int ts = flip ? (8 - t) : t;
  Wh[(t * I + i) * O + o] = (_Float16)(W[(o * I + i) * 9 + ts] * scale);
}

// ---------------------------------------------------------------------------
// Build modulated, zero-padded (and optionally zero-stuffed x2 for transposed
// conv) f16 input in CHANNELS-LAST layout: xmCL[b][py][px][I], Hp x Hp.
//   stride=1, off=1 : Hp = Hin+2      (normal conv, pad 1)
//   stride=2, off=2 : Hp = 2*Hin+3    (lhs-dilated by 2, pad 2)
// Channels-last makes K contiguous per pixel so the WMMA kernel can load
// operand fragments straight from global memory with b128 loads.
// ---------------------------------------------------------------------------
__global__ void k_modpad_cl(const float* __restrict__ x, const float* __restrict__ style,
                            _Float16* __restrict__ xmCL, int I, int Hin, int Hp,
                            int off, int stride) {
  int idx = blockIdx.x * 256 + threadIdx.x;
  if (idx >= 4 * I * Hp * Hp) return;
  int i = idx % I; int tmp = idx / I;
  int px = tmp % Hp; tmp /= Hp;
  int py = tmp % Hp; int b = tmp / Hp;
  float v = 0.f;
  int sy = py - off, sx = px - off;
  if (sy >= 0 && sx >= 0) {
    if (stride == 2) {
      if (((sy | sx) & 1) == 0) {
        sy >>= 1; sx >>= 1;
        if (sy < Hin && sx < Hin)
          v = x[((b * I + i) * Hin + sy) * Hin + sx] * style[b * I + i];
      }
    } else {
      if (sy < Hin && sx < Hin)
        v = x[((b * I + i) * Hin + sy) * Hin + sx] * style[b * I + i];
    }
  }
  xmCL[idx] = (_Float16)v;  // idx == ((b*Hp+py)*Hp+px)*I + i
}

// ---------------------------------------------------------------------------
// WMMA implicit-GEMM 3x3 conv, LDS-free. Input is channels-last padded f16
// (Hp = Ho+2); weights are [tap][I][O] f16. M = 4*Ho*Wo, N = O, K = 9 x I.
// Block = 256 threads (8 waves); macro-tile 128(M) x 64(N); each wave owns a
// 16x64 strip -> 4x v_wmma_f32_16x16x32_f16 per 32-wide K-step, f32 accum.
// Operand fragments are loaded directly from global per the ISA 16-bit
// layouts (A: lane = M row, lane-half splits K; B: lane = N col, lane-half splits K, elements step K by O),
// which channels-last makes 16B-aligned contiguous A loads; B is read per element. O must be a
// multiple of 64 (true for all layers here).
// flags: bit0 = demodulate, bit1 = +noise & fused leaky relu.
// ---------------------------------------------------------------------------
__global__ void __launch_bounds__(256) k_conv3(
    const _Float16* __restrict__ xmCL, const _Float16* __restrict__ Wh,
    const float* __restrict__ demod, const float* __restrict__ noise,
    const float* __restrict__ nwp, const float* __restrict__ ab,
    float* __restrict__ out, int I, int O, int Ho, int Wo, int flags) {
  const int tid = threadIdx.x;
  const int lane = tid & 31, wv = tid >> 5;
  const int half = lane >> 4;
  const int Hp = Ho + 2, Wp = Wo + 2;
  const int HW = Ho * Wo;
  const int M_total = 4 * HW;
  const int mtile = blockIdx.x * 128;
  const int otile = blockIdx.y * 64;

  // Per-lane A row (M) -> pixel; clamp OOB rows (masked in epilogue).
  int m = mtile + wv * 16 + (lane & 15);
  if (m >= M_total) m = M_total - 1;
  int b = m / HW; int rr = m - b * HW;
  int y = rr / Wo; int x = rr - y * Wo;
  const _Float16* aPix = xmCL + (size_t)((b * Hp + y) * Wp + x) * I;

  v8f acc[4] = {{}, {}, {}, {}};

  for (int t = 0; t < 9; ++t) {
    const int dy = t / 3, dx = t - dy * 3;
    const _Float16* aT = aPix + (dy * Wp + dx) * I + half * 8;
    const _Float16* bT = Wh + (size_t)(t * I + half * 8) * O + otile + (lane & 15);
    if (t < 8) {  // prefetch next tap's pixel row (global_prefetch_b8)
      int dy2 = (t + 1) / 3, dx2 = (t + 1) - dy2 * 3;
      __builtin_prefetch(aPix + (dy2 * Wp + dx2) * I, 0, 0);
    }
    for (int i0 = 0; i0 < I; i0 += 32) {
      v8h a0 = *(const v8h*)(aT + i0);        // K = i0 + half*8 + 0..7
      v8h a1 = *(const v8h*)(aT + i0 + 16);   // K = i0 + 16 + half*8 + 0..7
      v16h af = __builtin_shufflevector(a0, a1, 0, 1, 2, 3, 4, 5, 6, 7,
                                        8, 9, 10, 11, 12, 13, 14, 15);
      const _Float16* bRow = bT + (size_t)i0 * O;  // K = i0 + half*8 (+16), N = lane&15
#pragma unroll
      for (int g = 0; g < 4; ++g) {
        v16h bf;
        for (int s = 0; s < 16; ++s)
          bf[s] = bRow[(size_t)((s & 7) + (s >> 3) * 16) * O + g * 16];
        acc[g] = __builtin_amdgcn_wmma_f32_16x16x32_f16(
            false, af, false, bf, (short)0, acc[g], false, false);
      }
    }
  }

  // Epilogue. D layout: reg r, lanes 0-15 -> M=r, lanes 16-31 -> M=8+r; N=lane&15.
  const float nw = (flags & 2) ? nwp[0] : 0.f;
#pragma unroll
  for (int r = 0; r < 8; ++r) {
    int me = mtile + wv * 16 + half * 8 + r;
    if (me >= M_total) continue;
    int be = me / HW; int re = me - be * HW;
    int ye = re / Wo; int xe = re - ye * Wo;
#pragma unroll
    for (int g = 0; g < 4; ++g) {
      int o = otile + g * 16 + (lane & 15);
      float v = acc[g][r];
      if (flags & 1) v *= demod[be * O + o];
      if (flags & 2) {
        v += nw * noise[ye * Wo + xe];
        v += ab[o];
        v = (v > 0.f ? v : 0.2f * v) * SQRT2F;
      }
      out[((be * O + o) * Ho + ye) * Wo + xe] = v;
    }
  }
}

// ---------------------------------------------------------------------------
// Post-upconv blur (4-tap, gain 4: outer([1,3,3,1])/16), pad(1,1):
// (2n+1)^2 -> (2n)^2, fused with noise add + flrelu.
// ---------------------------------------------------------------------------
__global__ void k_blur_noise_act(const float* __restrict__ in,
                                 const float* __restrict__ noise,
                                 const float* __restrict__ nwp,
                                 const float* __restrict__ ab,
                                 float* __restrict__ out,
                                 int O, int Hin, int Hout) {
  int idx = blockIdx.x * 256 + threadIdx.x;
  if (idx >= 4 * O * Hout * Hout) return;
  int x = idx % Hout; int tmp = idx / Hout;
  int y = tmp % Hout; tmp /= Hout;
  int o = tmp % O;    int b = tmp / O;
  const float k1[4] = {1.f, 3.f, 3.f, 1.f};
  const float* base = in + (b * O + o) * Hin * Hin;
  float acc = 0.f;
  for (int p = 0; p < 4; ++p) {
    int u = y - 1 + p;
    if (u < 0 || u >= Hin) continue;
    for (int q = 0; q < 4; ++q) {
      int v = x - 1 + q;
      if (v < 0 || v >= Hin) continue;
      acc += k1[p] * k1[q] * (1.f / 16.f) * base[u * Hin + v];
    }
  }
  acc += nwp[0] * noise[y * Hout + x] + ab[o];
  out[idx] = (acc > 0.f ? acc : 0.2f * acc) * SQRT2F;
}

// ---------------------------------------------------------------------------
// toRGB: 1x1 modulated conv (no demod) + bias + optional skip upsample
// (zero-stuff x2, pad(2,1), gain-4 blur). N=3 so scalar dot is appropriate.
// ---------------------------------------------------------------------------
__global__ void k_rgb(const float* __restrict__ x, const float* __restrict__ style,
                      const float* __restrict__ W3, const float* __restrict__ bias,
                      const float* __restrict__ skipIn, float* __restrict__ skipOut,
                      int I, int H, int Hs) {
  int idx = blockIdx.x * 256 + threadIdx.x;
  if (idx >= 4 * 3 * H * H) return;
  int px = idx % H; int tmp = idx / H;
  int py = tmp % H; tmp /= H;
  int c = tmp % 3;  int b = tmp / 3;
  const float sc = rsqrtf((float)I);
  float acc = bias[c];
  const float* xb = x + (b * I) * H * H + py * H + px;
  const float* st = style + b * I;
  const float* wr = W3 + c * I;
  for (int i = 0; i < I; ++i) acc += xb[i * H * H] * st[i] * wr[i] * sc;
  if (skipIn) {
    const float k1[4] = {1.f, 3.f, 3.f, 1.f};
    const float* sb = skipIn + (b * 3 + c) * Hs * Hs;
    for (int p = 0; p < 4; ++p) {
      int u = py + p - 2;
      if (u < 0 || (u & 1)) continue;
      u >>= 1; if (u >= Hs) continue;
      for (int q = 0; q < 4; ++q) {
        int v = px + q - 2;
        if (v < 0 || (v & 1)) continue;
        v >>= 1; if (v >= Hs) continue;
        acc += k1[p] * k1[q] * (1.f / 16.f) * sb[u * Hs + v];
      }
    }
  }
  skipOut[idx] = acc;
}

// ---------------------------------------------------------------------------
// Tile learned const over batch.
// ---------------------------------------------------------------------------
__global__ void k_const(const float* __restrict__ cst, float* __restrict__ out) {
  int idx = blockIdx.x * 256 + threadIdx.x;
  if (idx >= 4 * 512 * 16) return;
  out[idx] = cst[idx % (512 * 16)];
}

// ---------------------------------------------------------------------------
static inline int ceil_div(int a, int b) { return (a + b - 1) / b; }

extern "C" void kernel_launch(void* const* d_in, const int* in_sizes, int n_in,
                              void* d_out, int out_size, void* d_ws, size_t ws_size,
                              hipStream_t stream) {
  (void)in_sizes; (void)n_in; (void)out_size; (void)ws_size;
  // ---- inputs (setup_inputs dict order, lists flattened per element) ----
  const float* z        = (const float*)d_in[0];
  const float* map_w    = (const float*)d_in[1];
  const float* map_b    = (const float*)d_in[2];
  const float* cst      = (const float*)d_in[3];
  const float* conv1_w  = (const float*)d_in[4];
  const float* conv1_mw = (const float*)d_in[5];
  const float* conv1_mb = (const float*)d_in[6];
  const float* conv1_nw = (const float*)d_in[7];
  const float* conv1_ab = (const float*)d_in[8];
  const float* rgb1_w   = (const float*)d_in[9];
  const float* rgb1_mw  = (const float*)d_in[10];
  const float* rgb1_mb  = (const float*)d_in[11];
  const float* rgb1_b   = (const float*)d_in[12];
  const float *up_w[6], *up_mw[6], *up_mb[6], *up_nw[6], *up_ab[6];
  const float *c_w[6], *c_mw[6], *c_mb[6], *c_nw[6], *c_ab[6];
  const float *rgb_w[6], *rgb_mw[6], *rgb_mb[6], *rgb_b[6];
  for (int s = 0; s < 6; ++s) {
    up_w[s]   = (const float*)d_in[13 + s];
    up_mw[s]  = (const float*)d_in[19 + s];
    up_mb[s]  = (const float*)d_in[25 + s];
    up_nw[s]  = (const float*)d_in[31 + s];
    up_ab[s]  = (const float*)d_in[37 + s];
    c_w[s]    = (const float*)d_in[43 + s];
    c_mw[s]   = (const float*)d_in[49 + s];
    c_mb[s]   = (const float*)d_in[55 + s];
    c_nw[s]   = (const float*)d_in[61 + s];
    c_ab[s]   = (const float*)d_in[67 + s];
    rgb_w[s]  = (const float*)d_in[73 + s];
    rgb_mw[s] = (const float*)d_in[79 + s];
    rgb_mb[s] = (const float*)d_in[85 + s];
    rgb_b[s]  = (const float*)d_in[91 + s];
  }
  const float* noi[13];
  for (int k = 0; k < 13; ++k) noi[k] = (const float*)d_in[97 + k];

  // ---- workspace carve-up ----
  char* ws = (char*)d_ws;
  size_t off = 0;
  float*    wlat  = (float*)(ws + off);    off += 16 << 10;
  float*    style = (float*)(ws + off);    off += 16 << 10;
  float*    demod = (float*)(ws + off);    off += 16 << 10;
  _Float16* wh    = (_Float16*)(ws + off); off += (size_t)5 << 20;   // 9*512*512 f16
  _Float16* xmCL  = (_Float16*)(ws + off); off += (size_t)160 << 20; // padded f16 input
  float*    act0  = (float*)(ws + off);    off += (size_t)136 << 20;
  float*    act1  = (float*)(ws + off);    off += (size_t)136 << 20;
  float*    tconv = (float*)(ws + off);    off += (size_t)140 << 20; // (2n+1)^2 pre-blur
  float*    skipA = (float*)(ws + off);    off += (size_t)4 << 20;
  float*    skipB = (float*)(ws + off);

  // ---- mapping network + const input ----
  k_mapping<<<1, 512, 0, stream>>>(z, map_w, map_b, wlat);
  k_const<<<ceil_div(4 * 512 * 16, 256), 256, 0, stream>>>(cst, act0);

  auto styled = [&](const float* xin, float* yout, const float* Wc,
                    const float* mwp, const float* mbp, const float* nwp,
                    const float* abp, const float* np, int I, int O, int n, int up) {
    k_style<<<ceil_div(4 * I, 256), 256, 0, stream>>>(wlat, mwp, mbp, style, I);
    k_demod<<<ceil_div(4 * O, 256), 256, 0, stream>>>(style, Wc, demod, I, O);
    float scale = 1.0f / sqrtf((float)(I * 9));
    k_wconv<<<ceil_div(O * I * 9, 256), 256, 0, stream>>>(Wc, wh, O, I, up, scale);
    if (!up) {
      int Hp = n + 2;
      k_modpad_cl<<<ceil_div(4 * I * Hp * Hp, 256), 256, 0, stream>>>(
          xin, style, xmCL, I, n, Hp, 1, 1);
      dim3 g(ceil_div(4 * n * n, 128), ceil_div(O, 64));
      k_conv3<<<g, 256, 0, stream>>>(xmCL, wh, demod, np, nwp, abp, yout,
                                     I, O, n, n, /*flags=*/3);
    } else {
      int Ho = 2 * n + 1, Hp = 2 * n + 3, Hout = 2 * n;
      k_modpad_cl<<<ceil_div(4 * I * Hp * Hp, 256), 256, 0, stream>>>(
          xin, style, xmCL, I, n, Hp, 2, 2);
      dim3 g(ceil_div(4 * Ho * Ho, 128), ceil_div(O, 64));
      k_conv3<<<g, 256, 0, stream>>>(xmCL, wh, demod, nullptr, nullptr, nullptr,
                                     tconv, I, O, Ho, Ho, /*flags=*/1);
      k_blur_noise_act<<<ceil_div(4 * O * Hout * Hout, 256), 256, 0, stream>>>(
          tconv, np, nwp, abp, yout, O, Ho, Hout);
    }
  };
  auto torgb = [&](const float* xin, const float* Wc, const float* mwp,
                   const float* mbp, const float* bias, const float* skipIn,
                   float* skipOut, int I, int H) {
    k_style<<<ceil_div(4 * I, 256), 256, 0, stream>>>(wlat, mwp, mbp, style, I);
    k_rgb<<<ceil_div(4 * 3 * H * H, 256), 256, 0, stream>>>(
        xin, style, Wc, bias, skipIn, skipOut, I, H, H / 2);
  };

  // ---- synthesis stack ----
  styled(act0, act1, conv1_w, conv1_mw, conv1_mb, conv1_nw, conv1_ab,
         noi[0], 512, 512, 4, /*up=*/0);
  torgb(act1, rgb1_w, rgb1_mw, rgb1_mb, rgb1_b, nullptr, skipA, 512, 4);

  const int chans[6] = {512, 512, 512, 512, 256, 128};
  float* cur = act1;
  float* oth = act0;
  float* skipPrev = skipA;
  float* skipNext = skipB;
  int ic = 512;
  for (int s = 0; s < 6; ++s) {
    int oc = chans[s];
    int n = 4 << s;   // input res
    int r = 8 << s;   // output res
    styled(cur, oth, up_w[s], up_mw[s], up_mb[s], up_nw[s], up_ab[s],
           noi[1 + 2 * s], ic, oc, n, /*up=*/1);
    styled(oth, cur, c_w[s], c_mw[s], c_mb[s], c_nw[s], c_ab[s],
           noi[2 + 2 * s], oc, oc, r, /*up=*/0);
    float* dst = (s == 5) ? (float*)d_out : skipNext;
    torgb(cur, rgb_w[s], rgb_mw[s], rgb_mb[s], rgb_b[s], skipPrev, dst, oc, r);
    skipNext = skipPrev;  // the buffer we just consumed is free next round
    skipPrev = dst;
    ic = oc;
  }
}
